// Block_86672440033530
// MI455X (gfx1250) — compile-verified, matrix operand order checked by reading, not run on hardware
//
#include <hip/hip_runtime.h>
#include <hip/hip_bf16.h>

// ---------------------------------------------------------------------------
// Types for CDNA5 WMMA (wave32): V_WMMA_F32_16X16X32_BF16
// ---------------------------------------------------------------------------
typedef __attribute__((ext_vector_type(8)))  float   v8f;
typedef __attribute__((ext_vector_type(16))) __bf16  v16bf;
typedef __attribute__((ext_vector_type(8)))  __bf16  v8bf;

union Frag16 { v16bf v; v8bf h[2]; };

static __device__ inline v8f vzero8() {
  v8f z = {0.f, 0.f, 0.f, 0.f, 0.f, 0.f, 0.f, 0.f};
  return z;
}

static __device__ inline v8f wmma_bf16(const Frag16& a, const Frag16& b, v8f c) {
  return __builtin_amdgcn_wmma_f32_16x16x32_bf16(
      /*neg_a=*/false, a.v, /*neg_b=*/false, b.v,
      /*c_mod=*/(short)0, c, /*reuse_a=*/false, /*reuse_b=*/false);
}

// Problem constants
#define MB_   4096   // B*S rows
#define DM_   1024   // d_model
#define DFF_  4096
#define SS_   2048   // sequence length
#define HH_   16
#define HD_   64

// ---------------------------------------------------------------------------
// Weight convert + transpose: W[K][N] fp32 -> Wt[N][K] bf16 (32x32 LDS tiles)
// ---------------------------------------------------------------------------
__global__ __launch_bounds__(256)
void wconv_t(const float* __restrict__ W, __bf16* __restrict__ Wt, int Kd, int Nd) {
  __shared__ float t[32][33];
  const int k0 = blockIdx.y * 32, n0 = blockIdx.x * 32;
  for (int i = threadIdx.y; i < 32; i += 8)
    t[i][threadIdx.x] = W[(size_t)(k0 + i) * Nd + n0 + threadIdx.x];
  __syncthreads();
  for (int i = threadIdx.y; i < 32; i += 8)
    Wt[(size_t)(n0 + i) * Kd + k0 + threadIdx.x] = (__bf16)t[threadIdx.x][i];
}

// ---------------------------------------------------------------------------
// WMMA GEMM: C[M,N] = A[M,K] * Bt[N,K]^T  (+bias, +residual, relu, f32/bf16 out)
// Block tile 128x128x64, 256 threads = 8 waves, each wave 64(M) x 32(N).
// 16 WMMAs per LDS stage (2 K-steps) to amortize barriers.
// store_mode 1 writes outB transposed per-head: [B,H,HD,S] (for V projection).
// ---------------------------------------------------------------------------
#define GBM 128
#define GBN 128
#define GBK 64
#define GLDS 72   // LDS row stride (elements); 144B rows keep b128 alignment

template <typename AT>
__global__ __launch_bounds__(256)
void gemm_wmma(const AT* __restrict__ A, const __bf16* __restrict__ Bt,
               int M, int N, int K,
               const float* __restrict__ bias, const float* __restrict__ residual,
               int relu, float* __restrict__ outF, __bf16* __restrict__ outB,
               int store_mode) {
  __shared__ __bf16 sA[GBM * GLDS];
  __shared__ __bf16 sB[GBN * GLDS];

  const int tid  = threadIdx.x;
  const int lane = tid & 31;
  const int wave = tid >> 5;
  const int wm   = wave >> 2;      // 0..1 -> 64 rows each
  const int wn   = wave & 3;       // 0..3 -> 32 cols each
  const int m0   = blockIdx.y * GBM;
  const int n0   = blockIdx.x * GBN;

  const int srow = tid >> 3;       // 0..31
  const int scol = (tid & 7) * 8;  // 0..56

  v8f acc[4][2];
#pragma unroll
  for (int i = 0; i < 4; ++i)
#pragma unroll
    for (int j = 0; j < 2; ++j) acc[i][j] = vzero8();

  const int ka = (lane < 16) ? 0 : 8;
  const int lc = lane & 15;

  for (int k0 = 0; k0 < K; k0 += GBK) {
    // stage A (convert to bf16) and B tiles into LDS
#pragma unroll
    for (int p = 0; p < 4; ++p) {
      int row = p * 32 + srow;
      const AT* src = A + (size_t)(m0 + row) * K + k0 + scol;
      v8bf tmp;
#pragma unroll
      for (int i = 0; i < 8; ++i) tmp[i] = (__bf16)(float)src[i];
      *(v8bf*)&sA[row * GLDS + scol] = tmp;
      if (k0 + GBK < K) __builtin_prefetch(src + GBK, 0, 1);  // global_prefetch_b8
    }
#pragma unroll
    for (int p = 0; p < 4; ++p) {
      int row = p * 32 + srow;
      const __bf16* src = Bt + (size_t)(n0 + row) * K + k0 + scol;
      *(v8bf*)&sB[row * GLDS + scol] = *(const v8bf*)src;
      if (k0 + GBK < K) __builtin_prefetch(src + GBK, 0, 1);
    }
    __syncthreads();

#pragma unroll
    for (int ks = 0; ks < 2; ++ks) {
      Frag16 fa[4], fb[2];
#pragma unroll
      for (int mt = 0; mt < 4; ++mt) {
        int r = wm * 64 + mt * 16 + lc;
        fa[mt].h[0] = *(const v8bf*)&sA[r * GLDS + ks * 32 + ka];
        fa[mt].h[1] = *(const v8bf*)&sA[r * GLDS + ks * 32 + ka + 16];
      }
#pragma unroll
      for (int nt = 0; nt < 2; ++nt) {
        int r = wn * 32 + nt * 16 + lc;
        fb[nt].h[0] = *(const v8bf*)&sB[r * GLDS + ks * 32 + ka];
        fb[nt].h[1] = *(const v8bf*)&sB[r * GLDS + ks * 32 + ka + 16];
      }
#pragma unroll
      for (int mt = 0; mt < 4; ++mt)
#pragma unroll
        for (int nt = 0; nt < 2; ++nt)
          acc[mt][nt] = wmma_bf16(fa[mt], fb[nt], acc[mt][nt]);
    }
    __syncthreads();
  }

  // epilogue (C layout: lane<16 -> M=r, lane>=16 -> M=r+8; N = lane&15)
  const int mh = (lane < 16) ? 0 : 8;
#pragma unroll
  for (int mt = 0; mt < 4; ++mt)
#pragma unroll
    for (int nt = 0; nt < 2; ++nt)
#pragma unroll
      for (int r = 0; r < 8; ++r) {
        int m = m0 + wm * 64 + mt * 16 + r + mh;
        int n = n0 + wn * 32 + nt * 16 + lc;
        float v = acc[mt][nt][r];
        if (bias) v += bias[n];
        if (residual) v += residual[(size_t)m * N + n];
        if (relu) v = fmaxf(v, 0.f);
        if (outF) outF[(size_t)m * N + n] = v;
        if (outB) {
          size_t idx;
          if (store_mode == 1) {  // V^T layout [B,H,HD,S]
            int b = m >> 11, s = m & (SS_ - 1);
            int h = n >> 6, hd = n & (HD_ - 1);
            idx = ((((size_t)b * HH_ + h) * HD_ + hd) << 11) + s;
          } else {
            idx = (size_t)m * N + n;
          }
          outB[idx] = (__bf16)v;
        }
      }
}

// ---------------------------------------------------------------------------
// Flash attention with BigBird mask.
// Grid: (B*H, S/128). Block 256 = 8 waves; each wave owns 16 query rows.
// Streams 64-key chunks: S = Q*K^T (WMMA) -> mask (LDS-staged) / scale ->
// online softmax (shfl_xor half-wave reductions) -> P staged via LDS ->
// O += P*V (WMMA against V^T tile).
// ---------------------------------------------------------------------------
__global__ __launch_bounds__(256)
void flash_attn(const __bf16* __restrict__ Q, const __bf16* __restrict__ Kb,
                const __bf16* __restrict__ Vt, const unsigned char* __restrict__ mask,
                __bf16* __restrict__ O) {
  __shared__ __bf16 sQ[128 * 72];
  __shared__ __bf16 sK[64 * 72];
  __shared__ __bf16 sV[64 * 72];                 // V^T tile: [hd][s]
  __shared__ __bf16 sP[8 * 16 * 72];             // per-wave P tiles [16][64]
  __shared__ __align__(16) unsigned char sM[128 * 64];  // mask tile [qrow][kcol]

  const int bh = blockIdx.x;
  const int b  = bh >> 4;
  const int h  = bh & 15;
  const int q0 = blockIdx.y * 128;
  const int tid = threadIdx.x, lane = tid & 31, wave = tid >> 5;
  const int lc = lane & 15;
  const int ka = (lane < 16) ? 0 : 8;
  const int mh = (lane < 16) ? 0 : 8;
  const size_t rowbase = (size_t)b * SS_;

  // Load Q tile (128 rows x 64 cols, bf16)
  {
    const int c8 = (tid & 7) * 8;
#pragma unroll
    for (int p = 0; p < 4; ++p) {
      int row = p * 32 + (tid >> 3);
      *(v8bf*)&sQ[row * 72 + c8] =
          *(const v8bf*)&Q[(rowbase + q0 + row) * DM_ + h * HD_ + c8];
    }
  }

  v8f o[4];
#pragma unroll
  for (int t = 0; t < 4; ++t) o[t] = vzero8();
  float mrun[8], lrun[8];
#pragma unroll
  for (int r = 0; r < 8; ++r) { mrun[r] = -1e30f; lrun[r] = 0.f; }

  for (int k0 = 0; k0 < SS_; k0 += 64) {
    __syncthreads();  // protect sK/sV/sM reuse (also orders initial sQ fill)
    {
      const int c8 = (tid & 7) * 8;
#pragma unroll
      for (int p = 0; p < 2; ++p) {
        int row = p * 32 + (tid >> 3);
        *(v8bf*)&sK[row * 72 + c8] =
            *(const v8bf*)&Kb[(rowbase + k0 + row) * DM_ + h * HD_ + c8];
        *(v8bf*)&sV[row * 72 + c8] =
            *(const v8bf*)&Vt[((size_t)bh * HD_ + row) * SS_ + k0 + c8];
      }
    }
    // stage mask tile: coalesced 16B loads (128 rows x 64 bytes)
    {
      const int mr = tid >> 2;          // 0..63
      const int mc = (tid & 3) * 16;    // 0,16,32,48
#pragma unroll
      for (int p = 0; p < 2; ++p) {
        int row = p * 64 + mr;
        *(uint4*)&sM[row * 64 + mc] =
            *(const uint4*)&mask[(size_t)(q0 + row) * SS_ + k0 + mc];
      }
    }
    __syncthreads();

    // scores: wave rows [wave*16, +16) x 64 key columns
    v8f sc[4];
#pragma unroll
    for (int nt = 0; nt < 4; ++nt) sc[nt] = vzero8();
#pragma unroll
    for (int ks = 0; ks < 2; ++ks) {
      Frag16 fq;
      int qr = wave * 16 + lc;
      fq.h[0] = *(const v8bf*)&sQ[qr * 72 + ks * 32 + ka];
      fq.h[1] = *(const v8bf*)&sQ[qr * 72 + ks * 32 + ka + 16];
#pragma unroll
      for (int nt = 0; nt < 4; ++nt) {
        Frag16 fk;
        int kr = nt * 16 + lc;
        fk.h[0] = *(const v8bf*)&sK[kr * 72 + ks * 32 + ka];
        fk.h[1] = *(const v8bf*)&sK[kr * 72 + ks * 32 + ka + 16];
        sc[nt] = wmma_bf16(fq, fk, sc[nt]);
      }
    }

    // mask + scale + online softmax (per accumulator row r)
    float alpha[8];
#pragma unroll
    for (int r = 0; r < 8; ++r) {
      const int lrow = wave * 16 + r + mh;  // local q row 0..127
      float vmax = -1e30f;
#pragma unroll
      for (int nt = 0; nt < 4; ++nt) {
        float s = sc[nt][r] * 0.125f;  // 1/sqrt(64)
        if (!sM[lrow * 64 + nt * 16 + lc]) s = -1e9f;
        sc[nt][r] = s;
        vmax = fmaxf(vmax, s);
      }
#pragma unroll
      for (int d = 1; d < 16; d <<= 1) vmax = fmaxf(vmax, __shfl_xor(vmax, d, 32));
      float mn = fmaxf(mrun[r], vmax);
      alpha[r] = __expf(mrun[r] - mn);
      float lsum = 0.f;
#pragma unroll
      for (int nt = 0; nt < 4; ++nt) {
        float p = __expf(sc[nt][r] - mn);
        sc[nt][r] = p;
        lsum += p;
      }
#pragma unroll
      for (int d = 1; d < 16; d <<= 1) lsum += __shfl_xor(lsum, d, 32);
      lrun[r] = lrun[r] * alpha[r] + lsum;
      mrun[r] = mn;
    }

    // rescale O
#pragma unroll
    for (int t = 0; t < 4; ++t)
#pragma unroll
      for (int r = 0; r < 8; ++r) o[t][r] *= alpha[r];

    // stage P (C layout -> A layout via per-wave LDS tile)
#pragma unroll
    for (int r = 0; r < 8; ++r) {
      int lrow = r + mh;
#pragma unroll
      for (int nt = 0; nt < 4; ++nt)
        sP[(wave * 16 + lrow) * 72 + nt * 16 + lc] = (__bf16)sc[nt][r];
    }

    // O += P(16x64) * V(64x64)  (B fragments from V^T tile)
#pragma unroll
    for (int ks = 0; ks < 2; ++ks) {
      Frag16 fp;
      int pr = wave * 16 + lc;
      fp.h[0] = *(const v8bf*)&sP[pr * 72 + ks * 32 + ka];
      fp.h[1] = *(const v8bf*)&sP[pr * 72 + ks * 32 + ka + 16];
#pragma unroll
      for (int nt = 0; nt < 4; ++nt) {
        Frag16 fv;
        int vr = nt * 16 + lc;
        fv.h[0] = *(const v8bf*)&sV[vr * 72 + ks * 32 + ka];
        fv.h[1] = *(const v8bf*)&sV[vr * 72 + ks * 32 + ka + 16];
        o[nt] = wmma_bf16(fp, fv, o[nt]);
      }
    }
  }

  // normalize + store attn output bf16 [4096,1024]
#pragma unroll
  for (int nt = 0; nt < 4; ++nt)
#pragma unroll
    for (int r = 0; r < 8; ++r) {
      int mrow = q0 + wave * 16 + r + mh;
      int col = h * HD_ + nt * 16 + lc;
      O[(rowbase + mrow) * DM_ + col] = (__bf16)(o[nt][r] / lrun[r]);
    }
}

// ---------------------------------------------------------------------------
// Row-vector x matrix: out[b][n] = sum_k x[b][k] * W[k][n] + bias[n]
// (degenerate cross-attention path; softmax over 1 key == 1)
// ---------------------------------------------------------------------------
__global__ __launch_bounds__(256)
void rowvec_matmul(const float* __restrict__ x, const float* __restrict__ W,
                   const float* __restrict__ bias, float* __restrict__ out,
                   int Kd, int Nd) {
  const int b = blockIdx.y;
  const int n = blockIdx.x * 256 + threadIdx.x;
  if (n >= Nd) return;
  float acc = bias ? bias[n] : 0.f;
  for (int k = 0; k < Kd; ++k) acc += x[b * Kd + k] * W[(size_t)k * Nd + n];
  out[b * Nd + n] = acc;
}

// ---------------------------------------------------------------------------
// LayerNorm over D=1024 (1 block/row, 256 threads). Optional per-batch row
// addend (cross-attn broadcast). Writes fp32 and optional bf16 copy.
// ---------------------------------------------------------------------------
__global__ __launch_bounds__(256)
void ln_kernel(const float* __restrict__ X, const float* __restrict__ rowadd,
               const float* __restrict__ g, const float* __restrict__ beta,
               float* __restrict__ outF, __bf16* __restrict__ outB) {
  __shared__ float wred[8];
  const int row = blockIdx.x;
  const int tid = threadIdx.x;
  const int b = row >> 11;  // 2048 rows per batch

  float xv[4];
  float s = 0.f;
#pragma unroll
  for (int i = 0; i < 4; ++i) {
    int c = tid + i * 256;
    float v = X[(size_t)row * DM_ + c];
    if (rowadd) v += rowadd[b * DM_ + c];
    xv[i] = v;
    s += v;
  }
#pragma unroll
  for (int d = 1; d < 32; d <<= 1) s += __shfl_xor(s, d, 32);
  if ((tid & 31) == 0) wred[tid >> 5] = s;
  __syncthreads();
  float tot = 0.f;
#pragma unroll
  for (int i = 0; i < 8; ++i) tot += wred[i];
  const float mean = tot * (1.f / DM_);
  __syncthreads();

  float s2 = 0.f;
#pragma unroll
  for (int i = 0; i < 4; ++i) {
    float d0 = xv[i] - mean;
    s2 += d0 * d0;
  }
#pragma unroll
  for (int d = 1; d < 32; d <<= 1) s2 += __shfl_xor(s2, d, 32);
  if ((tid & 31) == 0) wred[tid >> 5] = s2;
  __syncthreads();
  float tot2 = 0.f;
#pragma unroll
  for (int i = 0; i < 8; ++i) tot2 += wred[i];
  const float inv = rsqrtf(tot2 * (1.f / DM_) + 1e-5f);

#pragma unroll
  for (int i = 0; i < 4; ++i) {
    int c = tid + i * 256;
    float y = (xv[i] - mean) * inv * g[c] + beta[c];
    if (outF) outF[(size_t)row * DM_ + c] = y;
    if (outB) outB[(size_t)row * DM_ + c] = (__bf16)y;
  }
}

// ---------------------------------------------------------------------------
// Host launch
// ---------------------------------------------------------------------------
extern "C" void kernel_launch(void* const* d_in, const int* in_sizes, int n_in,
                              void* d_out, int out_size, void* d_ws, size_t ws_size,
                              hipStream_t stream) {
  (void)in_sizes; (void)n_in; (void)out_size; (void)ws_size;
  const float* cur_input  = (const float*)d_in[0];
  const float* classVec   = (const float*)d_in[1];
  const unsigned char* am = (const unsigned char*)d_in[2];
  const float* sa_wq = (const float*)d_in[3];  const float* sa_bq = (const float*)d_in[4];
  const float* sa_wk = (const float*)d_in[5];  const float* sa_bk = (const float*)d_in[6];
  const float* sa_wv = (const float*)d_in[7];  const float* sa_bv = (const float*)d_in[8];
  const float* sa_wo = (const float*)d_in[9];  const float* sa_bo = (const float*)d_in[10];
  const float* ln1_g = (const float*)d_in[11]; const float* ln1_b = (const float*)d_in[12];
  const float* ce_w  = (const float*)d_in[13]; const float* ce_b  = (const float*)d_in[14];
  // ca_wq/bq (15,16), ca_wk/bk (17,18) are mathematically dead (softmax over 1 key == 1)
  const float* ca_wv = (const float*)d_in[19]; const float* ca_bv = (const float*)d_in[20];
  const float* ca_wo = (const float*)d_in[21]; const float* ca_bo = (const float*)d_in[22];
  const float* ln2_g = (const float*)d_in[23]; const float* ln2_b = (const float*)d_in[24];
  const float* ff_w1 = (const float*)d_in[25]; const float* ff_b1 = (const float*)d_in[26];
  const float* ff_w2 = (const float*)d_in[27]; const float* ff_b2 = (const float*)d_in[28];
  const float* ln3_g = (const float*)d_in[29]; const float* ln3_b = (const float*)d_in[30];

  char* w = (char*)d_ws;
  size_t off = 0;
  auto alloc = [&](size_t bytes) -> char* {
    char* p = w + off;
    off = (off + bytes + 255) & ~(size_t)255;
    return p;
  };
  const size_t E = (size_t)MB_ * DM_;  // 4M elements
  __bf16* wq_t = (__bf16*)alloc((size_t)DM_ * DM_ * 2);
  __bf16* wk_t = (__bf16*)alloc((size_t)DM_ * DM_ * 2);
  __bf16* wv_t = (__bf16*)alloc((size_t)DM_ * DM_ * 2);
  __bf16* wo_t = (__bf16*)alloc((size_t)DM_ * DM_ * 2);
  __bf16* w1_t = (__bf16*)alloc((size_t)DM_ * DFF_ * 2);
  __bf16* w2_t = (__bf16*)alloc((size_t)DFF_ * DM_ * 2);
  __bf16* qb   = (__bf16*)alloc(E * 2);
  __bf16* kb   = (__bf16*)alloc(E * 2);
  __bf16* vtb  = (__bf16*)alloc(E * 2);      // [B,H,HD,S]
  __bf16* ao   = (__bf16*)alloc(E * 2);      // attn out bf16
  __bf16* hb   = (__bf16*)alloc(E * 2);      // h bf16
  __bf16* h2b  = (__bf16*)alloc(E * 2);      // h2 bf16
  __bf16* f1b  = (__bf16*)alloc((size_t)MB_ * DFF_ * 2);
  float*  sum1 = (float*)alloc(E * 4);       // reused as sum2
  float*  hbuf = (float*)alloc(E * 4);
  float*  h2buf= (float*)alloc(E * 4);
  float*  cvb  = (float*)alloc(2 * DM_ * 4);
  float*  vrow = (float*)alloc(2 * DM_ * 4);
  float*  carow= (float*)alloc(2 * DM_ * 4);

  dim3 tb(32, 8);
  wconv_t<<<dim3(DM_ / 32, DM_ / 32), tb, 0, stream>>>(sa_wq, wq_t, DM_, DM_);
  wconv_t<<<dim3(DM_ / 32, DM_ / 32), tb, 0, stream>>>(sa_wk, wk_t, DM_, DM_);
  wconv_t<<<dim3(DM_ / 32, DM_ / 32), tb, 0, stream>>>(sa_wv, wv_t, DM_, DM_);
  wconv_t<<<dim3(DM_ / 32, DM_ / 32), tb, 0, stream>>>(sa_wo, wo_t, DM_, DM_);
  wconv_t<<<dim3(DFF_ / 32, DM_ / 32), tb, 0, stream>>>(ff_w1, w1_t, DM_, DFF_);
  wconv_t<<<dim3(DM_ / 32, DFF_ / 32), tb, 0, stream>>>(ff_w2, w2_t, DFF_, DM_);

  const dim3 gD(DM_ / GBN, MB_ / GBM);    // N=1024 GEMMs
  const dim3 gF(DFF_ / GBN, MB_ / GBM);   // N=4096 GEMM
  // QKV projections (A fp32 from cur_input, converted to bf16 in LDS)
  gemm_wmma<float><<<gD, 256, 0, stream>>>(cur_input, wq_t, MB_, DM_, DM_,
      sa_bq, nullptr, 0, nullptr, qb, 0);
  gemm_wmma<float><<<gD, 256, 0, stream>>>(cur_input, wk_t, MB_, DM_, DM_,
      sa_bk, nullptr, 0, nullptr, kb, 0);
  gemm_wmma<float><<<gD, 256, 0, stream>>>(cur_input, wv_t, MB_, DM_, DM_,
      sa_bv, nullptr, 0, nullptr, vtb, 1);  // transposed per-head store

  flash_attn<<<dim3(2 * HH_, SS_ / 128), 256, 0, stream>>>(qb, kb, vtb, am, ao);

  // out projection + residual(cur_input) -> sum1; LN1 -> h (fp32 + bf16)
  gemm_wmma<__bf16><<<gD, 256, 0, stream>>>(ao, wo_t, MB_, DM_, DM_,
      sa_bo, cur_input, 0, sum1, nullptr, 0);
  ln_kernel<<<MB_, 256, 0, stream>>>(sum1, nullptr, ln1_g, ln1_b, hbuf, hb);

  // degenerate cross-attention: carow[b] = ((cls@ce_w+ce_b)@ca_wv+ca_bv)@ca_wo+ca_bo
  rowvec_matmul<<<dim3(DM_ / 256, 2), 256, 0, stream>>>(classVec, ce_w, ce_b, cvb, 10, DM_);
  rowvec_matmul<<<dim3(DM_ / 256, 2), 256, 0, stream>>>(cvb, ca_wv, ca_bv, vrow, DM_, DM_);
  rowvec_matmul<<<dim3(DM_ / 256, 2), 256, 0, stream>>>(vrow, ca_wo, ca_bo, carow, DM_, DM_);

  // h2 = LN(h + carow broadcast)
  ln_kernel<<<MB_, 256, 0, stream>>>(hbuf, carow, ln2_g, ln2_b, h2buf, h2b);

  // FFN: relu(h2@W1+b1)@W2+b2 + h2 -> LN3 -> out
  gemm_wmma<__bf16><<<gF, 256, 0, stream>>>(h2b, w1_t, MB_, DFF_, DM_,
      ff_b1, nullptr, 1, nullptr, f1b, 0);
  gemm_wmma<__bf16><<<gD, 256, 0, stream>>>(f1b, w2_t, MB_, DM_, DFF_,
      ff_b2, h2buf, 0, sum1, nullptr, 0);
  ln_kernel<<<MB_, 256, 0, stream>>>(sum1, nullptr, ln3_g, ln3_b, (float*)d_out, nullptr);
}
